// RecomGCL_82008105549987
// MI455X (gfx1250) — matrix operand order repaired and checked by reading, not compiled, not run on hardware
//
#include <hip/hip_runtime.h>
#include <math.h>

// ---------------- problem constants (from reference) ----------------
#define N_U 100000
#define N_I 100000
#define DIM 64
#define QQ 5
#define N_EDGE 2000000
#define NLAYER 2
#define BB 4096
#define PP 10
// TEMP=0.2 -> 1/TEMP = 5 ; LAM1=LAM2=0.2 ; DROPOUT=0.1 -> keep=0.9 ; EPS=0.1

typedef float v8f  __attribute__((ext_vector_type(8)));
typedef __bf16 v16bf __attribute__((ext_vector_type(16)));

// ---------------- helpers ----------------
__device__ __forceinline__ unsigned hash2(unsigned x, unsigned y) {
  unsigned h = x * 0x9E3779B9u + y * 0x85EBCA6Bu + 0x27220A95u;
  h ^= h >> 16; h *= 0x7FEB352Du;
  h ^= h >> 15; h *= 0x846CA68Bu;
  h ^= h >> 16;
  return h;
}
__device__ __forceinline__ float unif(unsigned x, unsigned y) {
  return (float)(hash2(x, y) >> 8) * (1.0f / 16777216.0f);
}
__device__ __forceinline__ float lrelu(float x) { return x >= 0.0f ? x : 0.5f * x; }
__device__ __forceinline__ float sgnf(float x) { return (x > 0.0f) ? 1.0f : ((x < 0.0f) ? -1.0f : 0.0f); }
__device__ __forceinline__ unsigned short f2bf(float f) {
  unsigned u = __float_as_uint(f);
  unsigned r = u + 0x7FFFu + ((u >> 16) & 1u);
  return (unsigned short)(r >> 16);
}

// ---------------- generic zero / init ----------------
__global__ void k_zero(float* __restrict__ p, size_t n) {
  size_t i = (size_t)blockIdx.x * blockDim.x + threadIdx.x;
  size_t stride = (size_t)gridDim.x * blockDim.x;
  for (; i < n; i += stride) p[i] = 0.0f;
}
__global__ void k_copy2(const float* __restrict__ src, float* __restrict__ d1,
                        float* __restrict__ d2, size_t n) {
  size_t i = (size_t)blockIdx.x * blockDim.x + threadIdx.x;
  size_t stride = (size_t)gridDim.x * blockDim.x;
  for (; i < n; i += stride) { float v = src[i]; d1[i] = v; d2[i] = v; }
}

// ---------------- SPMM: one wave per edge, 2 dims per lane, 8 atomics ----------------
__global__ void k_spmm(const float* __restrict__ vals, const int* __restrict__ rows,
                       const int* __restrict__ cols,
                       const float* __restrict__ Eu, const float* __restrict__ Ei,
                       float* __restrict__ Zu, float* __restrict__ Zdu,
                       float* __restrict__ Zi, float* __restrict__ Zdi,
                       int nE, unsigned salt) {
  int gid = blockIdx.x * blockDim.x + threadIdx.x;
  int wid = gid >> 5, lane = gid & 31;
  int nW = (gridDim.x * blockDim.x) >> 5;
  const float2* Eu2 = (const float2*)Eu;
  const float2* Ei2 = (const float2*)Ei;
  for (int e = wid; e < nE; e += nW) {
    // stream-prefetch the next edge record (edge arrays are the HBM-resident stream)
    if (e + nW < nE) {
      __builtin_prefetch(&rows[e + nW], 0, 1);
      __builtin_prefetch(&cols[e + nW], 0, 1);
      __builtin_prefetch(&vals[e + nW], 0, 1);
    }
    int r = rows[e], c = cols[e];
    float v = vals[e];
    // sparse dropout (two independent masks), keep=0.9
    float v1 = (unif(salt * 2u + 0u, (unsigned)e) < 0.9f) ? v * (1.0f / 0.9f) : 0.0f;
    float v2 = (unif(salt * 2u + 1u, (unsigned)e) < 0.9f) ? v * (1.0f / 0.9f) : 0.0f;
    float2 xi = Ei2[(size_t)c * 32 + lane];
    float2 xu = Eu2[(size_t)r * 32 + lane];
    float* zu  = Zu  + (size_t)r * DIM + lane * 2;
    float* zdu = Zdu + (size_t)r * DIM + lane * 2;
    float* zi  = Zi  + (size_t)c * DIM + lane * 2;
    float* zdi = Zdi + (size_t)c * DIM + lane * 2;
    atomicAdd(zu,     v  * xi.x); atomicAdd(zu + 1,  v  * xi.y);
    atomicAdd(zdu,    v1 * xi.x); atomicAdd(zdu + 1, v1 * xi.y);
    atomicAdd(zi,     v  * xu.x); atomicAdd(zi + 1,  v  * xu.y);
    atomicAdd(zdi,    v2 * xu.x); atomicAdd(zdi + 1, v2 * xu.y);
  }
}

// ---------------- T[q][d] = sum_n vt[q][n] * X[n][d]  (rank-5 reduction) ----------------
__global__ void k_qd_reduce(const float* __restrict__ vtm, const float* __restrict__ X,
                            float* __restrict__ T, int N) {
  int d = threadIdx.x;  // 64 threads
  int chunk = (N + gridDim.x - 1) / gridDim.x;
  int n0 = blockIdx.x * chunk;
  int n1 = n0 + chunk; if (n1 > N) n1 = N;
  float acc[QQ] = {0.f, 0.f, 0.f, 0.f, 0.f};
  for (int n = n0; n < n1; ++n) {
    float x = X[(size_t)n * DIM + d];
#pragma unroll
    for (int q = 0; q < QQ; ++q) acc[q] += vtm[(size_t)q * N + n] * x;
  }
#pragma unroll
  for (int q = 0; q < QQ; ++q) atomicAdd(&T[q * DIM + d], acc[q]);
}

// ---------------- act + noise + residual update (wave per row, 2 dims/lane) ----------------
__global__ void k_update(float* __restrict__ Zn, float* __restrict__ Zd,
                         float* __restrict__ Ep, float* __restrict__ Ea,
                         int N, unsigned salt) {
  int gid = blockIdx.x * blockDim.x + threadIdx.x;
  int wid = gid >> 5, lane = gid & 31;
  int nW = (gridDim.x * blockDim.x) >> 5;
  for (int row = wid; row < N; row += nW) {
    unsigned base = (unsigned)row * 64u;
    float n0 = unif(salt, base + lane * 2u);
    float n1 = unif(salt, base + lane * 2u + 1u);
    float ss = n0 * n0 + n1 * n1;
    ss += __shfl_xor(ss, 1, 32);  ss += __shfl_xor(ss, 2, 32);
    ss += __shfl_xor(ss, 4, 32);  ss += __shfl_xor(ss, 8, 32);
    ss += __shfl_xor(ss, 16, 32);
    float scl = 0.1f / fmaxf(sqrtf(ss), 1e-12f);  // EPS * l2n(noise)
    size_t idx = (size_t)row * DIM + lane * 2;
    float z0 = lrelu(Zn[idx]), z1 = lrelu(Zn[idx + 1]);
    z0 += sgnf(z0) * n0 * scl; z1 += sgnf(z1) * n1 * scl;
    Zn[idx] = z0; Zn[idx + 1] = z1;
    float d0 = lrelu(Zd[idx]), d1 = lrelu(Zd[idx + 1]);
    Zd[idx] = d0; Zd[idx + 1] = d1;
    float e0 = Ep[idx] + z0 + d0, e1 = Ep[idx + 1] + z1 + d1;
    Ep[idx] = e0; Ep[idx + 1] = e1;
    Ea[idx] += e0; Ea[idx + 1] += e1;
  }
}

// ---------------- gather batch rows + SVD-view G = act(s_mul[id] @ T) ----------------
__global__ void k_gather(const float* __restrict__ Z, const float* __restrict__ Zd,
                         const float* __restrict__ smul, const float* __restrict__ T,
                         const int* __restrict__ ids,
                         float* __restrict__ Zb, float* __restrict__ Zdb,
                         float* __restrict__ Gb) {
  int t = blockIdx.x * blockDim.x + threadIdx.x;
  if (t >= BB * DIM) return;
  int b = t >> 6, d = t & 63;
  int id = ids[b];
  Zb[t]  = Z[(size_t)id * DIM + d];
  Zdb[t] = Zd[(size_t)id * DIM + d];
  float a = 0.f;
#pragma unroll
  for (int q = 0; q < QQ; ++q) a += smul[(size_t)id * QQ + q] * T[q * DIM + d];
  Gb[t] = lrelu(a);
}

// ---------------- InfoNCE prep: normalize, hyper @ W, pos dot, pack bf16 ----------------
__global__ void k_nce_prep(const float* __restrict__ g_raw, const float* __restrict__ h_raw,
                           const float* __restrict__ W,
                           unsigned short* __restrict__ g16, unsigned short* __restrict__ h16,
                           float* __restrict__ posb, float* __restrict__ negb) {
  __shared__ float red[DIM];
  __shared__ float hn[DIM];
  int b = blockIdx.x, d = threadIdx.x;
  float g = g_raw[(size_t)b * DIM + d];
  float h = h_raw[(size_t)b * DIM + d];
  red[d] = g * g; __syncthreads();
  for (int s = 32; s > 0; s >>= 1) { if (d < s) red[d] += red[d + s]; __syncthreads(); }
  float ng = fmaxf(sqrtf(red[0]), 1e-12f); __syncthreads();
  red[d] = h * h; __syncthreads();
  for (int s = 32; s > 0; s >>= 1) { if (d < s) red[d] += red[d + s]; __syncthreads(); }
  float nh = fmaxf(sqrtf(red[0]), 1e-12f); __syncthreads();
  float gn = g / ng;
  g16[(size_t)b * DIM + d] = f2bf(gn);
  hn[d] = h / nh; __syncthreads();
  float acc = 0.f;
#pragma unroll 8
  for (int k = 0; k < DIM; ++k) acc += hn[k] * W[k * DIM + d];
  h16[(size_t)b * DIM + d] = f2bf(acc);
  // pos_score raw dot: sum_d gnn[b][d] * hyper[b][d]  (f32, matches reference exactly)
  __syncthreads();
  red[d] = gn * acc; __syncthreads();
  for (int s = 32; s > 0; s >>= 1) { if (d < s) red[d] += red[d + s]; __syncthreads(); }
  if (d == 0) { posb[b] = red[0]; negb[b] = 0.f; }
}

// ---------------- InfoNCE GEMM: S = gnn @ hyperT  (bf16 WMMA), exp + row-sum ----------------
// One wave per 16-tile strip along N. A operand register-resident (reused x16).
// Per tile the two k-step WMMAs are kept INDEPENDENT (both C=0, summed in f32 after)
// so there is no WMMA->WMMA RAW hazard; unroll capped at 2 to bound register
// liveness (no spills); B fragments double-buffered so exp/accum of tile `it`
// overlaps the loads of tile `it+1`. Single shuffle-reduce epilogue per strip.
#define STRIP 16
union bfrag { unsigned u[8]; v16bf v; };

__global__ void k_nce_gemm(const unsigned short* __restrict__ A16,
                           const unsigned short* __restrict__ B16,
                           float* __restrict__ negb) {
  int wid = (blockIdx.x * blockDim.x + threadIdx.x) >> 5;
  int lane = threadIdx.x & 31;
  int tm = wid >> 4;              // tile row 0..255
  int sc = wid & 15;              // strip index 0..15 (STRIP tiles each)
  int mr = tm << 4;
  int li = lane & 15, g = lane >> 4;
  const unsigned* Aw = (const unsigned*)A16;  // 32 dwords per 64-elem bf16 row
  const unsigned* Bw = (const unsigned*)B16;
  int arow = mr + li;
  // A k order used here (contiguous, same as B): lane group g holds K = 16*g + 2*r (pairs)
  bfrag ua0, ua1;
#pragma unroll
  for (int r = 0; r < 8; ++r) {
    int offA = g * 16 + 2 * r;
    ua0.u[r] = Aw[(size_t)arow * 32 + (offA >> 1)];
    ua1.u[r] = Aw[(size_t)arow * 32 + ((32 + offA) >> 1)];
  }
  // B k order used here (contiguous, same as A): lane group g holds K = 16*g + 2*r (pairs)
  bfrag ub0[2], ub1[2];
  int brow0 = (sc << 8) + li;     // first tile of strip: nc = sc*256
#pragma unroll
  for (int r = 0; r < 8; ++r) {
    int offB = g * 16 + 2 * r;
    ub0[0].u[r] = Bw[(size_t)brow0 * 32 + (offB >> 1)];
    ub1[0].u[r] = Bw[(size_t)brow0 * 32 + ((32 + offB) >> 1)];
  }
  float racc[8] = {0.f, 0.f, 0.f, 0.f, 0.f, 0.f, 0.f, 0.f};
#pragma unroll 2
  for (int it = 0; it < STRIP; ++it) {
    int cur = it & 1, nxt = cur ^ 1;
    if (it + 1 < STRIP) {          // prefetch next tile's B fragments
      int brow = (sc << 8) + ((it + 1) << 4) + li;
#pragma unroll
      for (int r = 0; r < 8; ++r) {
        int offB = g * 16 + 2 * r;
        ub0[nxt].u[r] = Bw[(size_t)brow * 32 + (offB >> 1)];
        ub1[nxt].u[r] = Bw[(size_t)brow * 32 + ((32 + offB) >> 1)];
      }
    }
    // two independent partial products (no WMMA->WMMA dependency), summed in f32
    v8f z = {};
    v8f c0 = __builtin_amdgcn_wmma_f32_16x16x32_bf16(false, ua0.v, false, ub0[cur].v,
                                                     (short)0, z, false, false);
    v8f c1 = __builtin_amdgcn_wmma_f32_16x16x32_bf16(false, ua1.v, false, ub1[cur].v,
                                                     (short)0, z, false, false);
#pragma unroll
    for (int r = 0; r < 8; ++r) racc[r] += __expf((c0[r] + c1[r]) * 5.0f);
  }
  // reduce partial row-sums across the 16 n-lanes of each group, once per strip
#pragma unroll
  for (int r = 0; r < 8; ++r) {
    float s = racc[r];
    s += __shfl_xor(s, 1, 32); s += __shfl_xor(s, 2, 32);
    s += __shfl_xor(s, 4, 32); s += __shfl_xor(s, 8, 32);
    if (li == 0) atomicAdd(&negb[mr + g * 8 + r], s);
  }
}

// ---------------- InfoNCE finalize: masked -log(pos/(neg+eps)+eps) ----------------
__global__ void k_nce_fin(const float* __restrict__ posb, const float* __restrict__ negb,
                          float* __restrict__ slot, unsigned salt) {
  int b = blockIdx.x * blockDim.x + threadIdx.x;
  float term = 0.f;
  if (b < BB) {
    float mask = (unif(salt, (unsigned)b) > 0.5f) ? 1.0f : 0.0f;
    float ps = __expf(posb[b] * 5.0f);
    term = -logf(ps / (negb[b] + 1e-8f) + 1e-8f) * mask;
  }
  term += __shfl_xor(term, 1, 32);  term += __shfl_xor(term, 2, 32);
  term += __shfl_xor(term, 4, 32);  term += __shfl_xor(term, 8, 32);
  term += __shfl_xor(term, 16, 32);
  if ((threadIdx.x & 31) == 0) atomicAdd(slot, term);
}

// ---------------- BPR + hard-margin loss: one wave per batch element ----------------
__global__ void k_bpr(const float* __restrict__ Eu, const float* __restrict__ Ei,
                      const int* __restrict__ uids, const int* __restrict__ posI,
                      const int* __restrict__ negI, float* __restrict__ slot) {
  int gid = blockIdx.x * blockDim.x + threadIdx.x;
  int wid = gid >> 5, lane = gid & 31;
  if (wid >= BB) return;
  const float2* Eu2 = (const float2*)Eu;
  const float2* Ei2 = (const float2*)Ei;
  float2 u = Eu2[(size_t)uids[wid] * 32 + lane];
  float posmin = 1e30f, negmax = -1e30f, bpr = 0.f;
#pragma unroll
  for (int p = 0; p < PP; ++p) {
    float2 xp = Ei2[(size_t)posI[wid * PP + p] * 32 + lane];
    float2 xn = Ei2[(size_t)negI[wid * PP + p] * 32 + lane];
    float sp = u.x * xp.x + u.y * xp.y;
    float sn = u.x * xn.x + u.y * xn.y;
    for (int m = 1; m < 32; m <<= 1) { sp += __shfl_xor(sp, m, 32); sn += __shfl_xor(sn, m, 32); }
    posmin = fminf(posmin, sp); negmax = fmaxf(negmax, sn);
    bpr += fmaxf(1.0f - sp + sn, 0.0f);
  }
  if (lane == 0) {
    float diff = negmax - posmin;
    float hard = (diff > 0.005f) ? 10.0f * diff : 0.0f;
    atomicAdd(slot, (bpr + hard) * (1.0f / (float)BB));
  }
}

// ---------------- final combine ----------------
__global__ void k_combine(const float* __restrict__ sc, float* __restrict__ out) {
  if (blockIdx.x == 0 && threadIdx.x == 0) {
    float n = sc[0], d = sc[1], nn = sc[2], dd = sc[3], nd = sc[4], r = sc[5];
    out[0] = r + 0.2f * (n + d) + 0.2f * (nn + dd + nd);
    out[1] = r;
    out[2] = d;
    out[3] = n;
  }
}

// ==================================================================
extern "C" void kernel_launch(void* const* d_in, const int* in_sizes, int n_in,
                              void* d_out, int out_size, void* d_ws, size_t ws_size,
                              hipStream_t stream) {
  (void)in_sizes; (void)n_in; (void)out_size; (void)ws_size;
  const float* E_u_0   = (const float*)d_in[0];
  const float* E_i_0   = (const float*)d_in[1];
  const float* adj_vals= (const float*)d_in[2];
  const float* u_mul_s = (const float*)d_in[3];
  const float* v_mul_s = (const float*)d_in[4];
  const float* ut      = (const float*)d_in[5];
  const float* vt      = (const float*)d_in[6];
  const float* W_nn    = (const float*)d_in[7];
  const float* W_dd    = (const float*)d_in[8];
  const float* W_ng    = (const float*)d_in[9];
  const float* W_dg    = (const float*)d_in[10];
  const float* W_nd    = (const float*)d_in[11];
  const int* adj_rows  = (const int*)d_in[12];
  const int* adj_cols  = (const int*)d_in[13];
  const int* uids      = (const int*)d_in[14];
  const int* iids      = (const int*)d_in[15];
  const int* posI      = (const int*)d_in[16];
  const int* negI      = (const int*)d_in[17];

  // ---- workspace layout (~220 MB; each table is L2-resident on its own) ----
  char* ws = (char*)d_ws;
  size_t off = 0;
  auto alloc = [&](size_t nfloats) -> float* {
    float* p = (float*)(ws + off); off += nfloats * sizeof(float); return p;
  };
  const size_t ND = (size_t)N_U * DIM;
  float* Eu_prev = alloc(ND);  float* Ei_prev = alloc(ND);
  float* Eu_acc  = alloc(ND);  float* Ei_acc  = alloc(ND);
  float* Zu  = alloc(ND);  float* Zdu = alloc(ND);   // contiguous block of 4
  float* Zi  = alloc(ND);  float* Zdi = alloc(ND);
  const size_t BD = (size_t)BB * DIM;
  float* Zu_b[2]  = {alloc(BD), alloc(BD)};
  float* Zdu_b[2] = {alloc(BD), alloc(BD)};
  float* Zi_b[2]  = {alloc(BD), alloc(BD)};
  float* Zdi_b[2] = {alloc(BD), alloc(BD)};
  float* Gu_b[2]  = {alloc(BD), alloc(BD)};
  float* Gi_b[2]  = {alloc(BD), alloc(BD)};
  unsigned short* g16 = (unsigned short*)alloc(BD / 2);
  unsigned short* h16 = (unsigned short*)alloc(BD / 2);
  float* posb = alloc(BB);
  float* negb = alloc(BB);
  float* T_u  = alloc(QQ * DIM);   // contiguous pair
  float* T_i  = alloc(QQ * DIM);
  float* scalars = alloc(8);       // [n, d, nn, dd, nd, r, pad, pad]

  // ---- init ----
  k_zero <<<1, 64, 0, stream>>>(scalars, 8);
  k_copy2<<<4096, 256, 0, stream>>>(E_u_0, Eu_prev, Eu_acc, ND);
  k_copy2<<<4096, 256, 0, stream>>>(E_i_0, Ei_prev, Ei_acc, ND);

  // ---- propagation layers ----
  for (int l = 0; l < NLAYER; ++l) {
    k_zero<<<8192, 256, 0, stream>>>(Zu, 4 * ND);          // Zu,Zdu,Zi,Zdi contiguous
    k_zero<<<1, 256, 0, stream>>>(T_u, 2 * QQ * DIM);      // T_u,T_i contiguous
    k_spmm<<<2048, 256, 0, stream>>>(adj_vals, adj_rows, adj_cols, Eu_prev, Ei_prev,
                                     Zu, Zdu, Zi, Zdi, N_EDGE, 0x1000u + (unsigned)l);
    k_qd_reduce<<<512, 64, 0, stream>>>(vt, Ei_prev, T_u, N_I);
    k_qd_reduce<<<512, 64, 0, stream>>>(ut, Eu_prev, T_i, N_U);
    k_update<<<2048, 256, 0, stream>>>(Zu, Zdu, Eu_prev, Eu_acc, N_U, 0x2000u + (unsigned)l);
    k_update<<<2048, 256, 0, stream>>>(Zi, Zdi, Ei_prev, Ei_acc, N_I, 0x3000u + (unsigned)l);
    k_gather<<<(BB * DIM) / 256, 256, 0, stream>>>(Zu, Zdu, u_mul_s, T_u, uids,
                                                   Zu_b[l], Zdu_b[l], Gu_b[l]);
    k_gather<<<(BB * DIM) / 256, 256, 0, stream>>>(Zi, Zdi, v_mul_s, T_i, iids,
                                                   Zi_b[l], Zdi_b[l], Gi_b[l]);
  }

  // ---- 16 InfoNCE terms ----
  const size_t WSZ = (size_t)DIM * DIM;
  struct Call { const float* g; const float* h; const float* W; int slot; };
  Call calls[16] = {
    {Zu_b[0],  Gu_b[0],  W_ng,       0}, {Zu_b[1],  Gu_b[1],  W_ng + WSZ, 0},
    {Zi_b[0],  Gi_b[0],  W_ng,       0}, {Zi_b[1],  Gi_b[1],  W_ng + WSZ, 0},
    {Zu_b[0],  Zu_b[1],  W_nn,       2}, {Zi_b[0],  Zi_b[1],  W_nn,       2},
    {Zdu_b[0], Gu_b[0],  W_dg,       1}, {Zdu_b[1], Gu_b[1],  W_dg + WSZ, 1},
    {Zdi_b[0], Gi_b[0],  W_dg,       1}, {Zdi_b[1], Gi_b[1],  W_dg + WSZ, 1},
    {Zdu_b[0], Zdu_b[1], W_dd,       3}, {Zdi_b[0], Zdi_b[1], W_dd,       3},
    {Zu_b[0],  Zdu_b[0], W_nd,       4}, {Zu_b[1],  Zdu_b[1], W_nd + WSZ, 4},
    {Zi_b[0],  Zdi_b[0], W_nd,       4}, {Zi_b[1],  Zdi_b[1], W_nd + WSZ, 4},
  };
  // waves = 256 tile-rows x 16 strips = 4096 -> 512 blocks of 256 threads
  const int gemmBlocks = (256 * 16 * 32) / 256;
  for (int ci = 0; ci < 16; ++ci) {
    k_nce_prep<<<BB, DIM, 0, stream>>>(calls[ci].g, calls[ci].h, calls[ci].W,
                                       g16, h16, posb, negb);
    k_nce_gemm<<<gemmBlocks, 256, 0, stream>>>(g16, h16, negb);
    k_nce_fin<<<BB / 256, 256, 0, stream>>>(posb, negb, &scalars[calls[ci].slot],
                                            0xABCD0000u + (unsigned)ci);
  }

  // ---- BPR / hard-margin ranking loss ----
  k_bpr<<<(BB * 32) / 256, 256, 0, stream>>>(Eu_acc, Ei_acc, uids, posI, negI, &scalars[5]);

  // ---- combine ----
  k_combine<<<1, 64, 0, stream>>>(scalars, (float*)d_out);
}
